// MultiHeadAttention_498216206833
// MI455X (gfx1250) — matrix operand order repaired and checked by reading, not compiled, not run on hardware
//
#include <hip/hip_runtime.h>

#define DEVINL __device__ __forceinline__

typedef __attribute__((ext_vector_type(16))) __bf16 v16bf;
typedef __attribute__((ext_vector_type(8)))  __bf16 v8bf;
typedef __attribute__((ext_vector_type(8)))  float  v8f;

#if defined(__HIP_DEVICE_COMPILE__) && __has_builtin(__builtin_amdgcn_tensor_load_to_lds)
#define USE_TDM 1
typedef __attribute__((ext_vector_type(4))) unsigned int u32x4;
typedef __attribute__((ext_vector_type(8))) int i32x8;
typedef __attribute__((ext_vector_type(4))) int i32x4;
#else
#define USE_TDM 0
#endif

constexpr int B_   = 2;
constexpr int T_   = 2048;
constexpr int DIN  = 2048;
constexpr int DM   = 2048;
constexpr int H_   = 16;
constexpr int DH   = 128;
constexpr int NQKV = 3 * DM;      // 6144
constexpr int MROWS = B_ * T_;    // 4096

DEVINL unsigned short f2bf(float f) {
  union { float f; unsigned u; } v; v.f = f;
  unsigned r = v.u + 0x7FFFu + ((v.u >> 16) & 1u);
  return (unsigned short)(r >> 16);
}
DEVINL float bf2f(unsigned short h) {
  union { unsigned u; float f; } v; v.u = ((unsigned)h) << 16; return v.f;
}

// Build a 16-element bf16 fragment from two 16B-aligned 8-element chunks
// (matches CDNA5 WMMA per-lane K split across lane halves).
DEVINL v16bf ldfrag(const unsigned short* p0, const unsigned short* p1) {
  v8bf a = *(const v8bf*)p0;
  v8bf b = *(const v8bf*)p1;
  return __builtin_shufflevector(a, b, 0,1,2,3,4,5,6,7,8,9,10,11,12,13,14,15);
}

DEVINL v8f wmma_bf16(v16bf a, v16bf b, v8f c) {
  return __builtin_amdgcn_wmma_f32_16x16x32_bf16(
      /*neg_a=*/false, a, /*neg_b=*/false, b,
      /*c_mod=*/(short)0, c, /*reuse_a=*/false, /*reuse_b=*/false);
}

#if USE_TDM
// Low 32 bits of a generic pointer to LDS == raw LDS byte offset
// (flat->LDS mapping truncates to addr[31:0], ISA 10.2).
DEVINL unsigned lds_addr_of(const void* p) {
  return (unsigned)(unsigned long long)(uintptr_t)p;
}

// 2D bf16 tile DMA: d1_rows rows of d0_elems bf16 elements, global row stride
// row_stride_elems, written to LDS with pad_amt(+1 code) DWORD padding after
// every 2^(pad_int+1) DWORDs, matching padded __shared__ row strides.
// D# per CDNA5 ISA ch.8: group0 {count=1, lds_addr, global_addr57, type=2},
// group1 {data_size=2B, pad cfg, tensor/tile dims, dim0 stride}, groups2/3=0.
// This toolchain's builtin takes 6 args (extra i32x8 group before cpol).
DEVINL void tdm_load_2d(unsigned lds_addr, const void* gptr,
                        unsigned d0_elems, unsigned d1_rows,
                        unsigned row_stride_elems,
                        unsigned pad_int, unsigned pad_amt) {
  const unsigned long long ga = (unsigned long long)(uintptr_t)gptr;
  u32x4 g0 = { 1u,                                   // count=1 (valid user D#)
               lds_addr,                              // lds_addr[31:0]
               (unsigned)ga,                          // global_addr[31:0]
               (unsigned)((ga >> 32) & 0x01FFFFFFu)   // global_addr[56:32]
                   | 0x80000000u };                   // type=2 ("image")
  i32x8 g1 = { (int)((1u << 16)                      // data_size: 2 bytes
                   | (1u << 20)                      // pad_enable
                   | (pad_int << 22)                 // pad_interval code
                   | (pad_amt << 25)),               // pad_amount code
               (int)(d0_elems << 16),                // tensor_dim0[15:0]
               (int)(d1_rows << 16),                 // tensor_dim1[15:0]
               (int)(d0_elems << 16),                // tile_dim0
               (int)d1_rows,                         // tile_dim1 (tile_dim2=0)
               (int)row_stride_elems,                // tensor_dim0_stride lo32
               0, 0 };
  i32x4 gz4 = { 0, 0, 0, 0 };
  i32x8 gz8 = { 0, 0, 0, 0, 0, 0, 0, 0 };
  __builtin_amdgcn_tensor_load_to_lds(g0, g1, gz4, gz4, gz8, 0);
}
#endif

// ---------------------------------------------------------------------------
// Kernel 1: qkv = x @ W_qkv + b_qkv, scattered as bf16 into (B,H,T,DH) q/k/v.
// Block tile 128x128, K-step 32, 8 waves (4x2 wave grid, 2x4 WMMA frags/wave).
// ---------------------------------------------------------------------------
__global__ __launch_bounds__(256) void qkv_gemm_kernel(
    const float* __restrict__ x, const float* __restrict__ W,
    const float* __restrict__ bias,
    unsigned short* __restrict__ q, unsigned short* __restrict__ k,
    unsigned short* __restrict__ v) {
  __shared__ unsigned short As[128][40];  // [m][k]
  __shared__ unsigned short Bs[128][40];  // [n][k] (transposed for B-frags)
  const int tid  = threadIdx.x;
  const int lane = tid & 31, w = tid >> 5;
  const int wm = w & 3, wn = w >> 2;
  const int m0 = blockIdx.y * 128, n0 = blockIdx.x * 128;
  const int half8 = (lane >> 4) * 8;

  v8f acc[2][4];
#pragma unroll
  for (int i = 0; i < 2; ++i)
#pragma unroll
    for (int j = 0; j < 4; ++j) acc[i][j] = {};

  for (int k0 = 0; k0 < DIN; k0 += 32) {
    // Prefetch next K-slab of both streams into cache (global_prefetch_b8).
    if (k0 + 32 < DIN) {
      __builtin_prefetch(x + (size_t)(m0 + (tid >> 3)) * DIN + (k0 + 32) + (tid & 7) * 4, 0, 1);
      __builtin_prefetch(W + (size_t)(k0 + 32 + (tid >> 5)) * NQKV + n0 + (tid & 31) * 4, 0, 1);
    }
    // A tile: 128x32 f32 -> bf16
#pragma unroll
    for (int i = 0; i < 4; ++i) {
      int lin = tid + i * 256;
      int row = lin >> 3, c4 = lin & 7;
      const float4 a4 = *(const float4*)(x + (size_t)(m0 + row) * DIN + k0 + c4 * 4);
      ushort4 pk; pk.x = f2bf(a4.x); pk.y = f2bf(a4.y);
      pk.z = f2bf(a4.z); pk.w = f2bf(a4.w);
      *(ushort4*)&As[row][c4 * 4] = pk;
    }
    // B tile: 32x128 f32 -> bf16, transposed
#pragma unroll
    for (int i = 0; i < 4; ++i) {
      int lin = tid + i * 256;
      int kr = lin >> 5, c4 = lin & 31;
      const float4 b4 = *(const float4*)(W + (size_t)(k0 + kr) * NQKV + n0 + c4 * 4);
      Bs[c4 * 4 + 0][kr] = f2bf(b4.x);
      Bs[c4 * 4 + 1][kr] = f2bf(b4.y);
      Bs[c4 * 4 + 2][kr] = f2bf(b4.z);
      Bs[c4 * 4 + 3][kr] = f2bf(b4.w);
    }
    __syncthreads();
#pragma unroll
    for (int mi = 0; mi < 2; ++mi) {
      const int ar = 32 * wm + 16 * mi + (lane & 15);
      v16bf af = ldfrag(&As[ar][half8], &As[ar][half8 + 16]);
#pragma unroll
      for (int ni = 0; ni < 4; ++ni) {
        const int bn = 64 * wn + 16 * ni + (lane & 15);
        v16bf bf = ldfrag(&Bs[bn][half8], &Bs[bn][half8 + 16]);
        acc[mi][ni] = wmma_bf16(af, bf, acc[mi][ni]);
      }
    }
    __syncthreads();
  }

  // Epilogue: n0 is 128-aligned so this whole tile is one of q/k/v of head h.
  const int blk = n0 >> 7;
  const int h = blk / 3, which = blk % 3;
  unsigned short* dst = (which == 0) ? q : ((which == 1) ? k : v);
#pragma unroll
  for (int ni = 0; ni < 4; ++ni) {
    const int col = n0 + 64 * wn + 16 * ni + (lane & 15);
    const int d = col & 127;
    const float bb = bias[col];
#pragma unroll
    for (int mi = 0; mi < 2; ++mi) {
#pragma unroll
      for (int r = 0; r < 8; ++r) {
        const int gm = m0 + 32 * wm + 16 * mi + r + half8;
        const int bb_i = gm >> 11;          // / T_
        const int t = gm & (T_ - 1);
        dst[((size_t)(bb_i * H_ + h) * T_ + t) * DH + d] =
            f2bf(acc[mi][ni][r] + bb);
      }
    }
  }
}

// ---------------------------------------------------------------------------
// Kernel 2: in-place RoPE on q and k (bf16, layout (B,H,T,DH)).
// ---------------------------------------------------------------------------
__global__ __launch_bounds__(256) void rope_kernel(unsigned short* __restrict__ q,
                                                   unsigned short* __restrict__ k) {
  const int idx = blockIdx.x * 256 + threadIdx.x;   // over B*H*T*64
  const int d = idx & 63;
  const int t = (idx >> 6) & (T_ - 1);
  const int bh = idx >> 17;                         // / (64*T_)
  const size_t base = ((size_t)bh * T_ + t) * DH;
  // inv_freq = 10000^(-(2d)/128) = exp(-(2d/128)*ln(10000))
  const float inv = __expf((float)(2 * d) * (-9.2103403719761836f / 128.0f));
  float sv, cv;
  __sincosf((float)t * inv, &sv, &cv);
  {
    float x1 = bf2f(q[base + d]), x2 = bf2f(q[base + d + 64]);
    q[base + d]      = f2bf(x1 * cv - x2 * sv);
    q[base + d + 64] = f2bf(x2 * cv + x1 * sv);
  }
  {
    float x1 = bf2f(k[base + d]), x2 = bf2f(k[base + d + 64]);
    k[base + d]      = f2bf(x1 * cv - x2 * sv);
    k[base + d + 64] = f2bf(x2 * cv + x1 * sv);
  }
}

// ---------------------------------------------------------------------------
// Kernel 3: flash-style causal attention per (b,h) with 128-query blocks and
// 64-key iterations. Each wave owns 16 full score rows -> wave-local softmax.
// K-tile staged by the Tensor Data Mover (pad_interval=64DW, pad_amount=4DW
// reproduces Kb's 272B row stride); V transposed cooperatively in parallel.
// ---------------------------------------------------------------------------
__global__ __launch_bounds__(256) void attn_kernel(
    const unsigned short* __restrict__ q, const unsigned short* __restrict__ k,
    const unsigned short* __restrict__ v, unsigned short* __restrict__ o) {
  __shared__ unsigned short Kb[64][136];   // [key][d]   (B-frags of Q@K^T)
  __shared__ unsigned short Vt[128][72];   // [d][key]   (B-frags of P@V)
  __shared__ unsigned short Ps[128][72];   // [row][key] (A-frags of P@V)
  const int tid  = threadIdx.x;
  const int lane = tid & 31, w = tid >> 5;   // 8 waves: each owns 16 rows
  const int qi = blockIdx.x;
  const int bh = blockIdx.y;
  const size_t base = (size_t)bh * T_ * DH;
  const int half8 = (lane >> 4) * 8;
  const float scale = 0.08838834764831845f;   // 1/sqrt(128)

  // Persistent Q fragments: rows qi*128 + 16*w, full DH (4 K-steps of 32).
  v16bf qf[4];
  {
    const int qrow = qi * 128 + 16 * w + (lane & 15);
    const unsigned short* qp = q + base + (size_t)qrow * DH;
#pragma unroll
    for (int kk = 0; kk < 4; ++kk) {
      const int d0 = 32 * kk + half8;
      qf[kk] = ldfrag(qp + d0, qp + d0 + 16);
    }
  }

  v8f Oc[8];
  float mrow[8], lrow[8];
#pragma unroll
  for (int i = 0; i < 8; ++i) { Oc[i] = {}; mrow[i] = -3.0e38f; lrow[i] = 0.0f; }

  const int jmax = 2 * qi + 1;
  for (int j = 0; j <= jmax; ++j) {
    __syncthreads();
#if USE_TDM
    // Async DMA of the 64x128 bf16 K tile into padded LDS rows.
    if (w == 0) {
      tdm_load_2d(lds_addr_of(&Kb[0][0]),
                  k + base + (size_t)(j * 64) * DH,
                  /*d0_elems=*/128, /*d1_rows=*/64, /*row_stride=*/DH,
                  /*pad_int: 64 DW*/5, /*pad_amt: 4 DW*/3);
    }
#endif
    // Cooperative load: V block transposed (and K block when TDM unavailable).
#pragma unroll
    for (int i = 0; i < 4; ++i) {
      int lin = tid + i * 256;
      int row = lin >> 4, c = lin & 15;
      int key = j * 64 + row;
#if !USE_TDM
      *(uint4*)&Kb[row][c * 8] =
          *(const uint4*)(k + base + (size_t)key * DH + c * 8);
#endif
      uint4 vv = *(const uint4*)(v + base + (size_t)key * DH + c * 8);
      const unsigned short* ss = (const unsigned short*)&vv;
#pragma unroll
      for (int e = 0; e < 8; ++e) Vt[c * 8 + e][row] = ss[e];
    }
#if USE_TDM
    if (w == 0) __builtin_amdgcn_s_wait_tensorcnt(0);
#endif
    __syncthreads();

    // S = Q @ K^T : this wave computes rows [16w,16w+16) x all 64 keys.
    v8f S[4];
#pragma unroll
    for (int ni = 0; ni < 4; ++ni) S[ni] = {};
#pragma unroll
    for (int kk = 0; kk < 4; ++kk) {
#pragma unroll
      for (int ni = 0; ni < 4; ++ni) {
        const int key = 16 * ni + (lane & 15);
        v16bf bf = ldfrag(&Kb[key][32 * kk + half8],
                          &Kb[key][32 * kk + half8 + 16]);
        S[ni] = wmma_bf16(qf[kk], bf, S[ni]);
      }
    }

    // Scale + causal mask + online softmax (row stats are wave-local).
#pragma unroll
    for (int r = 0; r < 8; ++r) {
      const int grow = qi * 128 + 16 * w + r + half8;
      float p[4];
      float mx = -3.0e38f;
#pragma unroll
      for (int ni = 0; ni < 4; ++ni) {
        float s = S[ni][r] * scale;
        const int col = j * 64 + 16 * ni + (lane & 15);
        if (col > grow) s = -3.0e38f;
        p[ni] = s;
        mx = fmaxf(mx, s);
      }
#pragma unroll
      for (int off = 1; off < 16; off <<= 1)
        mx = fmaxf(mx, __shfl_xor(mx, off));
      const float mnew = fmaxf(mrow[r], mx);
      const float corr = __expf(mrow[r] - mnew);
      float rs = 0.0f;
#pragma unroll
      for (int ni = 0; ni < 4; ++ni) {
        p[ni] = __expf(p[ni] - mnew);
        rs += p[ni];
      }
#pragma unroll
      for (int off = 1; off < 16; off <<= 1)
        rs += __shfl_xor(rs, off);
      lrow[r] = lrow[r] * corr + rs;
      mrow[r] = mnew;
#pragma unroll
      for (int oni = 0; oni < 8; ++oni) Oc[oni][r] *= corr;
      const int rl = 16 * w + r + half8;
#pragma unroll
      for (int ni = 0; ni < 4; ++ni)
        Ps[rl][16 * ni + (lane & 15)] = f2bf(p[ni]);
    }
    __syncthreads();

    // O += P @ V : rows [16w,16w+16) x all 128 dims, K-dim = 64 keys.
#pragma unroll
    for (int kk = 0; kk < 2; ++kk) {
      const int pr = 16 * w + (lane & 15);
      v16bf af = ldfrag(&Ps[pr][32 * kk + half8],
                        &Ps[pr][32 * kk + half8 + 16]);
#pragma unroll
      for (int oni = 0; oni < 8; ++oni) {
        const int dcol = 16 * oni + (lane & 15);
        v16bf bf = ldfrag(&Vt[dcol][32 * kk + half8],
                          &Vt[dcol][32 * kk + half8 + 16]);
        Oc[oni] = wmma_bf16(af, bf, Oc[oni]);
      }
    }
  }

  // Normalize and write O as bf16 (B,H,T,DH).
#pragma unroll
  for (int r = 0; r < 8; ++r) {
    const float inv = 1.0f / lrow[r];
    const int grow = qi * 128 + 16 * w + r + half8;
#pragma unroll
    for (int oni = 0; oni < 8; ++oni) {
      const int dcol = 16 * oni + (lane & 15);
      o[base + (size_t)grow * DH + dcol] = f2bf(Oc[oni][r] * inv);
    }
  }
}

// ---------------------------------------------------------------------------
// Kernel 4: out = concat_heads(O) @ W_out + b_out  (f32 output).
// A tile (raw bf16 gather) staged by the TDM: 128 rows x 64B, global row
// stride 256B -> LDS 80B rows (pad_interval=16DW, pad_amount=4DW).
// ---------------------------------------------------------------------------
__global__ __launch_bounds__(256) void out_gemm_kernel(
    const unsigned short* __restrict__ o, const float* __restrict__ W,
    const float* __restrict__ bias, float* __restrict__ out) {
  __shared__ unsigned short As[128][40];
  __shared__ unsigned short Bs[128][40];
  const int tid  = threadIdx.x;
  const int lane = tid & 31, w = tid >> 5;
  const int wm = w & 3, wn = w >> 2;
  const int m0 = blockIdx.y * 128, n0 = blockIdx.x * 128;
  const int half8 = (lane >> 4) * 8;
  const int bb0 = m0 >> 11;            // batch of this row block
  const int t0 = m0 & (T_ - 1);        // first t of this row block

  v8f acc[2][4];
#pragma unroll
  for (int i = 0; i < 2; ++i)
#pragma unroll
    for (int j = 0; j < 4; ++j) acc[i][j] = {};

  for (int k0 = 0; k0 < DM; k0 += 32) {
    const int h = k0 >> 7;        // which head this 32-wide K-slab lives in
    const int dd = k0 & 127;
    if (k0 + 32 < DM) {
      __builtin_prefetch(W + (size_t)(k0 + 32 + (tid >> 5)) * DM + n0 + (tid & 31) * 4, 0, 1);
    }
#if USE_TDM
    if (w == 0) {
      tdm_load_2d(lds_addr_of(&As[0][0]),
                  o + ((size_t)(bb0 * H_ + h) * T_ + t0) * DH + dd,
                  /*d0_elems=*/32, /*d1_rows=*/128, /*row_stride=*/DH,
                  /*pad_int: 16 DW*/3, /*pad_amt: 4 DW*/3);
    }
#else
    // A tile: gather head-interleaved bf16 O (already bf16: raw copy)
#pragma unroll
    for (int i = 0; i < 2; ++i) {
      int lin = tid + i * 256;
      int row = lin >> 2, c = lin & 3;
      const int gm = m0 + row;
      const int bb_i = gm >> 11;
      const int t = gm & (T_ - 1);
      *(uint4*)&As[row][c * 8] =
          *(const uint4*)(o + ((size_t)(bb_i * H_ + h) * T_ + t) * DH + dd + c * 8);
    }
#endif
    // B tile: W_out 32x128 f32 -> bf16 transposed
#pragma unroll
    for (int i = 0; i < 4; ++i) {
      int lin = tid + i * 256;
      int kr = lin >> 5, c4 = lin & 31;
      const float4 b4 = *(const float4*)(W + (size_t)(k0 + kr) * DM + n0 + c4 * 4);
      Bs[c4 * 4 + 0][kr] = f2bf(b4.x);
      Bs[c4 * 4 + 1][kr] = f2bf(b4.y);
      Bs[c4 * 4 + 2][kr] = f2bf(b4.z);
      Bs[c4 * 4 + 3][kr] = f2bf(b4.w);
    }
#if USE_TDM
    if (w == 0) __builtin_amdgcn_s_wait_tensorcnt(0);
#endif
    __syncthreads();
#pragma unroll
    for (int mi = 0; mi < 2; ++mi) {
      const int ar = 32 * wm + 16 * mi + (lane & 15);
      v16bf af = ldfrag(&As[ar][half8], &As[ar][half8 + 16]);
#pragma unroll
      for (int ni = 0; ni < 4; ++ni) {
        const int bn = 64 * wn + 16 * ni + (lane & 15);
        v16bf bf = ldfrag(&Bs[bn][half8], &Bs[bn][half8 + 16]);
        acc[mi][ni] = wmma_bf16(af, bf, acc[mi][ni]);
      }
    }
    __syncthreads();
  }

#pragma unroll
  for (int ni = 0; ni < 4; ++ni) {
    const int col = n0 + 64 * wn + 16 * ni + (lane & 15);
    const float bb = bias[col];
#pragma unroll
    for (int mi = 0; mi < 2; ++mi) {
#pragma unroll
      for (int r = 0; r < 8; ++r) {
        const int gm = m0 + 32 * wm + 16 * mi + r + half8;
        out[(size_t)gm * DM + col] = acc[mi][ni][r] + bb;
      }
    }
  }
}

// ---------------------------------------------------------------------------
extern "C" void kernel_launch(void* const* d_in, const int* in_sizes, int n_in,
                              void* d_out, int out_size, void* d_ws, size_t ws_size,
                              hipStream_t stream) {
  (void)in_sizes; (void)n_in; (void)out_size; (void)ws_size;
  const float* x    = (const float*)d_in[0];
  const float* Wqkv = (const float*)d_in[1];
  const float* bqkv = (const float*)d_in[2];
  const float* Wout = (const float*)d_in[3];
  const float* bout = (const float*)d_in[4];
  // d_in[5] (mask) unused: causal mask is computed arithmetically.

  const size_t HTD = (size_t)B_ * H_ * T_ * DH;   // 8,388,608 elements
  unsigned short* q = (unsigned short*)d_ws;
  unsigned short* k = q + HTD;
  unsigned short* v = k + HTD;
  unsigned short* o = v + HTD;
  float* out = (float*)d_out;

  dim3 blk(256);
  qkv_gemm_kernel<<<dim3(NQKV / 128, MROWS / 128), blk, 0, stream>>>(
      x, Wqkv, bqkv, q, k, v);
  rope_kernel<<<(B_ * H_ * T_ * 64) / 256, blk, 0, stream>>>(q, k);
  attn_kernel<<<dim3(T_ / 128, B_ * H_), blk, 0, stream>>>(q, k, v, o);
  out_gemm_kernel<<<dim3(DM / 128, MROWS / 128), blk, 0, stream>>>(
      o, Wout, bout, out);
}
